// HierachicalRNN_91190745628890
// MI455X (gfx1250) — compile-verified, matrix operand order checked by reading, not run on hardware
//
#include <hip/hip_runtime.h>
#include <math.h>

// ---------------------------------------------------------------------------
// HierarchicalRNN forward for MI455X (gfx1250, wave32).
// All heavy GEMMs run on v_wmma_f32_16x16x32_bf16 (f32 accumulate).
//
// Model dims: V=32000 E=512 H=256 B=32 NH=8 NL=8 S=32; G=NH*NL=64, T=S=32.
// Input order (setup_inputs dict, params flattened in insertion order):
//  0 added_code(i32) 1 removed_code(i32) 2 hid_state_hunk 3 hid_state_sent
//  4 hid_state_word 5 embed
//  6..16  word_{Wih_f,Whh_f,bih_f,bhh_f,Wih_b,Whh_b,bih_b,bhh_b,Wa,ba,Wc}
//  17..27 sent_{ same }
//  28 Ws 29 bs 30 W1 31 b1 32 W2 33 b2 34 Wv 35 bv
// Output: (32, 3*512+4) f32.
// ---------------------------------------------------------------------------

typedef __bf16 bf16;
typedef __attribute__((ext_vector_type(16))) __bf16 v16bf;
typedef __attribute__((ext_vector_type(8)))  __bf16 v8bf;
typedef __attribute__((ext_vector_type(8)))  float  v8f;

__device__ __forceinline__ v8f zero_v8f() {
  v8f z = {0.f, 0.f, 0.f, 0.f, 0.f, 0.f, 0.f, 0.f};
  return z;
}

__device__ __forceinline__ v8f wmma_bf16(v16bf a, v16bf b, v8f c) {
  return __builtin_amdgcn_wmma_f32_16x16x32_bf16(false, a, false, b, (short)0, c,
                                                 false, false);
}

__device__ __forceinline__ unsigned pack2bf(float a, float b) {
  union { bf16 h[2]; unsigned u; } u2;
  u2.h[0] = (bf16)a;
  u2.h[1] = (bf16)b;
  return u2.u;
}

// A-fragment (16x32 bf16) from a row-major bf16 buffer (usually LDS).
// ISA 7.12.2: lane l<16 holds row M=l, K {0..7,16..23}; lane l>=16 row M=l-16,
// K {8..15,24..31}.  Two 16-byte loads per lane.
__device__ __forceinline__ v16bf load_a_frag(const bf16* buf, int rowlen,
                                             int row0, int kt, int lane) {
  int row = row0 + (lane & 15);
  int kb  = kt * 32 + ((lane & 16) ? 8 : 0);
  const bf16* p = buf + row * rowlen + kb;
  v8bf lo = *(const v8bf*)p;
  v8bf hi = *(const v8bf*)(p + 16);
  v16bf a;
#pragma unroll
  for (int i = 0; i < 8; ++i) { a[i] = lo[i]; a[i + 8] = hi[i]; }
  return a;
}

// B-fragment from pre-packed weights: packed[((kt*NT + nt)*32 + lane)*16].
__device__ __forceinline__ v16bf load_b_frag(const bf16* packed, int NT, int kt,
                                             int nt, int lane) {
  return *(const v16bf*)(packed + (((size_t)(kt * NT + nt)) * 32 + lane) * 16);
}

__device__ __forceinline__ float sigmoidf_(float x) {
  return 1.0f / (1.0f + expf(-x));
}

// ---------------------------------------------------------------------------
// Pack W (f32, row-major [N][K]) into bf16 WMMA B-fragments. B[k][n]=W[n][k].
// grid = (K/32)*(N/16), block = 32 (one lane group per tile).
// ---------------------------------------------------------------------------
__global__ void pack_b_kernel(const float* __restrict__ W, bf16* __restrict__ dst,
                              int K, int N) {
  int tile = blockIdx.x;
  int lane = threadIdx.x;
  int NT = N >> 4;
  int kt = tile / NT, nt = tile % NT;
  int col = nt * 16 + (lane & 15);
  int kb  = kt * 32 + ((lane & 16) ? 8 : 0);
  bf16* o = dst + ((size_t)tile * 32 + lane) * 16;
  const float* src = W + (size_t)col * K + kb;
#pragma unroll
  for (int i = 0; i < 8; ++i) o[i] = (bf16)src[i];
#pragma unroll
  for (int i = 0; i < 8; ++i) o[8 + i] = (bf16)src[16 + i];
}

// ---------------------------------------------------------------------------
// Embedding gather + f32->bf16, fully vectorized (float4 in, uint2 out).
// Row = (g*32 + t)*32 + n, g = h*8 + l; faithful index: token = x[t,h,l,n]
// (the (b,s)->(s,b) reshape is identity because b==s==32).
// grid = 65536 rows, block = 128 (each thread converts 4 floats).
// ---------------------------------------------------------------------------
__global__ void __launch_bounds__(128) gather_embed_kernel(
    const int* __restrict__ tok, const float* __restrict__ embed,
    bf16* __restrict__ xg) {
  int row = blockIdx.x;
  int g = row >> 10, t = (row >> 5) & 31, n = row & 31;
  int h = g >> 3, l = g & 7;
  int id = tok[((t * 8 + h) * 8 + l) * 32 + n];
  const float4* src = (const float4*)(embed + (size_t)id * 512);
  uint2* dst = (uint2*)(xg + (size_t)row * 512);
  float4 v = src[threadIdx.x];
  uint2 o;
  o.x = pack2bf(v.x, v.y);
  o.y = pack2bf(v.z, v.w);
  dst[threadIdx.x] = o;
}

// ---------------------------------------------------------------------------
// Bidirectional GRU scan.  blockIdx.x = g*2 + dir. 256 threads (8 waves),
// each wave owns 2 column-tiles of H.  Per step:
// gates = [x_t ; h] @ [Wih ; Whh].T (+biases), GRU elementwise.
// h double-buffered bf16 in LDS; weights streamed from L2-resident packed
// fragments; x-tile staged into LDS with b128 copies.
// ---------------------------------------------------------------------------
__global__ void __launch_bounds__(256) bigru_scan_kernel(
    const bf16* __restrict__ xg,       // [G][T][32][E]
    const float* __restrict__ h0,      // [2][32][256]
    const bf16* __restrict__ wih_f, const bf16* __restrict__ whh_f,
    const float* __restrict__ bih_f, const float* __restrict__ bhh_f,
    const bf16* __restrict__ wih_b, const bf16* __restrict__ whh_b,
    const float* __restrict__ bih_b, const float* __restrict__ bhh_b,
    bf16* __restrict__ out,            // [G][T][32][512]
    int T, int E) {
  __shared__ __align__(16) bf16 xbuf[32 * 512];
  __shared__ __align__(16) bf16 hbuf[2][32 * 256];

  const int g = blockIdx.x >> 1;
  const int dir = blockIdx.x & 1;
  const int tid = threadIdx.x;
  const int wave = tid >> 5;
  const int lane = tid & 31;

  const bf16*  Wih = dir ? wih_b : wih_f;
  const bf16*  Whh = dir ? whh_b : whh_f;
  const float* bih = dir ? bih_b : bih_f;
  const float* bhh = dir ? bhh_b : bhh_f;

  // init h from hid_state (vectorized f32x2 -> bf16x2)
  {
    const float2* s2 = (const float2*)(h0 + (size_t)dir * 32 * 256);
    unsigned* d1 = (unsigned*)hbuf[0];
    for (int i = tid; i < 32 * 256 / 2; i += 256) {
      float2 v = s2[i];
      d1[i] = pack2bf(v.x, v.y);
    }
  }
  __syncthreads();

  const int KTE = E >> 5;
  const int NV4 = (32 * E) >> 3;  // uint4 elements in the x tile
  for (int t = 0; t < T; ++t) {
    const int tsrc = dir ? (T - 1 - t) : t;
    const uint4* xsrc = (const uint4*)(xg + ((size_t)(g * T + tsrc)) * 32 * E);
    uint4* xdst = (uint4*)xbuf;
    for (int i = tid; i < NV4; i += 256) xdst[i] = xsrc[i];
    __syncthreads();

    const int cur = t & 1, nxt = cur ^ 1;
    for (int jt = wave; jt < 16; jt += 8) {  // column tile of H (uniform/wave)
      v8f Ar[2], Az[2], Anx[2], Anh[2];
#pragma unroll
      for (int m = 0; m < 2; ++m) {
        Ar[m] = zero_v8f(); Az[m] = zero_v8f();
        Anx[m] = zero_v8f(); Anh[m] = zero_v8f();
      }
      // input projection: K over E
      for (int kt = 0; kt < KTE; ++kt) {
        v16bf a0 = load_a_frag(xbuf, E, 0, kt, lane);
        v16bf a1 = load_a_frag(xbuf, E, 16, kt, lane);
        v16bf br = load_b_frag(Wih, 48, kt, jt, lane);
        v16bf bz = load_b_frag(Wih, 48, kt, 16 + jt, lane);
        v16bf bn = load_b_frag(Wih, 48, kt, 32 + jt, lane);
        Ar[0] = wmma_bf16(a0, br, Ar[0]);  Ar[1] = wmma_bf16(a1, br, Ar[1]);
        Az[0] = wmma_bf16(a0, bz, Az[0]);  Az[1] = wmma_bf16(a1, bz, Az[1]);
        Anx[0] = wmma_bf16(a0, bn, Anx[0]); Anx[1] = wmma_bf16(a1, bn, Anx[1]);
      }
      // hidden projection: K over H=256
      for (int kt = 0; kt < 8; ++kt) {
        v16bf a0 = load_a_frag(hbuf[cur], 256, 0, kt, lane);
        v16bf a1 = load_a_frag(hbuf[cur], 256, 16, kt, lane);
        v16bf br = load_b_frag(Whh, 48, kt, jt, lane);
        v16bf bz = load_b_frag(Whh, 48, kt, 16 + jt, lane);
        v16bf bn = load_b_frag(Whh, 48, kt, 32 + jt, lane);
        Ar[0] = wmma_bf16(a0, br, Ar[0]);  Ar[1] = wmma_bf16(a1, br, Ar[1]);
        Az[0] = wmma_bf16(a0, bz, Az[0]);  Az[1] = wmma_bf16(a1, bz, Az[1]);
        Anh[0] = wmma_bf16(a0, bn, Anh[0]); Anh[1] = wmma_bf16(a1, bn, Anh[1]);
      }
      // GRU elementwise (C/D layout: vgpr e -> M=e (+8 for lanes>=16), N=lane&15)
      const int col = jt * 16 + (lane & 15);
      const float br_ = bih[col] + bhh[col];
      const float bz_ = bih[256 + col] + bhh[256 + col];
      const float bxn = bih[512 + col];
      const float bhn = bhh[512 + col];
      const int tout = dir ? (T - 1 - t) : t;
      bf16* orow = out + ((size_t)(g * T + tout)) * 32 * 512 + dir * 256 + col;
#pragma unroll
      for (int m = 0; m < 2; ++m) {
#pragma unroll
        for (int e = 0; e < 8; ++e) {
          int row = m * 16 + e + ((lane & 16) ? 8 : 0);
          float r  = sigmoidf_(Ar[m][e] + br_);
          float z  = sigmoidf_(Az[m][e] + bz_);
          float nn = tanhf(Anx[m][e] + bxn + r * (Anh[m][e] + bhn));
          float hold = (float)hbuf[cur][row * 256 + col];
          float hnew = (1.0f - z) * nn + z * hold;
          hbuf[nxt][row * 256 + col] = (bf16)hnew;
          orow[(size_t)row * 512]    = (bf16)hnew;
        }
      }
    }
    __syncthreads();
  }
}

// ---------------------------------------------------------------------------
// Fused attention scoring: scores = ((src @ Wa.T) + ba) @ Wc.T, 32 rows/block.
// Intermediate ann kept in LDS as bf16.  grid = R/32, block = 256 (8 waves,
// 4 column-tiles each).
// ---------------------------------------------------------------------------
__global__ void __launch_bounds__(256) attn_scores_kernel(
    const bf16* __restrict__ src,   // [R][512]
    const bf16* __restrict__ wa_p, const float* __restrict__ ba,
    const bf16* __restrict__ wc_p,
    bf16* __restrict__ scores) {    // [R][512]
  __shared__ __align__(16) bf16 abuf[32 * 512];
  __shared__ __align__(16) bf16 annbuf[32 * 512];
  const int tid = threadIdx.x, wave = tid >> 5, lane = tid & 31;
  const size_t rowbase = (size_t)blockIdx.x * 32;

  {
    const uint4* s4 = (const uint4*)(src + rowbase * 512);
    uint4* d4 = (uint4*)abuf;
    for (int i = tid; i < (32 * 512) >> 3; i += 256) d4[i] = s4[i];
  }
  __syncthreads();

  for (int jt = wave; jt < 32; jt += 8) {
    v8f acc0 = zero_v8f(), acc1 = zero_v8f();
    for (int kt = 0; kt < 16; ++kt) {
      v16bf a0 = load_a_frag(abuf, 512, 0, kt, lane);
      v16bf a1 = load_a_frag(abuf, 512, 16, kt, lane);
      v16bf b  = load_b_frag(wa_p, 32, kt, jt, lane);
      acc0 = wmma_bf16(a0, b, acc0);
      acc1 = wmma_bf16(a1, b, acc1);
    }
    int col = jt * 16 + (lane & 15);
    float bb = ba[col];
#pragma unroll
    for (int e = 0; e < 8; ++e) {
      int r0 = e + ((lane & 16) ? 8 : 0);
      annbuf[r0 * 512 + col]        = (bf16)(acc0[e] + bb);
      annbuf[(r0 + 16) * 512 + col] = (bf16)(acc1[e] + bb);
    }
  }
  __syncthreads();

  for (int jt = wave; jt < 32; jt += 8) {
    v8f acc0 = zero_v8f(), acc1 = zero_v8f();
    for (int kt = 0; kt < 16; ++kt) {
      v16bf a0 = load_a_frag(annbuf, 512, 0, kt, lane);
      v16bf a1 = load_a_frag(annbuf, 512, 16, kt, lane);
      v16bf b  = load_b_frag(wc_p, 32, kt, jt, lane);
      acc0 = wmma_bf16(a0, b, acc0);
      acc1 = wmma_bf16(a1, b, acc1);
    }
    int col = jt * 16 + (lane & 15);
#pragma unroll
    for (int e = 0; e < 8; ++e) {
      int r0 = e + ((lane & 16) ? 8 : 0);
      scores[(rowbase + r0) * 512 + col]      = (bf16)acc0[e];
      scores[(rowbase + r0 + 16) * 512 + col] = (bf16)acc1[e];
    }
  }
}

// ---------------------------------------------------------------------------
// softmax over batch axis (faithful to reference) + Sum_t attn*out.
// grid = G, block = 256.  pooled (32x512 f32) lives in LDS.
// ---------------------------------------------------------------------------
__global__ void __launch_bounds__(256) softmax_pool_kernel(
    const bf16* __restrict__ scores,  // [G][T][32][512]
    const bf16* __restrict__ vals,    // [G][T][32][512]
    float* __restrict__ dst,          // [G][32][512]
    bf16* __restrict__ dst_bf,        // optional bf16 copy
    int T) {
  __shared__ float pooled[32 * 512];
  const int g = blockIdx.x, tid = threadIdx.x;
  for (int i = tid; i < 32 * 512; i += 256) pooled[i] = 0.f;
  __syncthreads();
  for (int t = 0; t < T; ++t) {
    const size_t base = ((size_t)(g * T + t)) * 32 * 512;
    for (int c = tid; c < 512; c += 256) {
      float sv[32];
      float mx = -3.4e38f;
#pragma unroll
      for (int n = 0; n < 32; ++n) {
        sv[n] = (float)scores[base + (size_t)n * 512 + c];
        mx = fmaxf(mx, sv[n]);
      }
      float s = 0.f;
#pragma unroll
      for (int n = 0; n < 32; ++n) { sv[n] = expf(sv[n] - mx); s += sv[n]; }
      float inv = 1.0f / s;
#pragma unroll
      for (int n = 0; n < 32; ++n)
        pooled[n * 512 + c] += sv[n] * inv * (float)vals[base + (size_t)n * 512 + c];
    }
  }
  __syncthreads();
  const size_t ob = (size_t)g * 32 * 512;
  for (int i = tid; i < 32 * 512; i += 256) {
    dst[ob + i] = pooled[i];
    if (dst_bf) dst_bf[ob + i] = (bf16)pooled[i];
  }
}

// mean over NH=8 hunks -> (32,512)
__global__ void mean8_kernel(const float* __restrict__ hunks,
                             float* __restrict__ vec) {
  int i = blockIdx.x * 256 + threadIdx.x;  // 0..16383
  float s = 0.f;
#pragma unroll
  for (int h = 0; h < 8; ++h) s += hunks[h * 32 * 512 + i];
  vec[i] = s * 0.125f;
}

// Head part 1: sub, mul, nn_out = relu([r,a]@Ws.T+bs), tmp = r@W{1,2}.T+b{1,2}
__global__ void head1_kernel(const float* __restrict__ a, const float* __restrict__ r,
                             const float* __restrict__ Ws, const float* __restrict__ bs,
                             const float* __restrict__ W1, const float* __restrict__ b1,
                             const float* __restrict__ W2, const float* __restrict__ b2,
                             float* __restrict__ tmp1, float* __restrict__ tmp2,
                             float* __restrict__ out) {
  int i = blockIdx.x * 256 + threadIdx.x;  // 0..16383
  int n = i >> 9, c = i & 511;
  const float* ar = a + n * 512;
  const float* rr = r + n * 512;
  float av = ar[c], rv = rr[c];
  float* orow = out + (size_t)n * 1540;
  orow[c]       = av - rv;
  orow[512 + c] = av * rv;
  float s = bs[c], t1 = b1[c], t2 = b2[c];
  const float* wsr = Ws + (size_t)c * 1024;
  const float* w1r = W1 + (size_t)c * 512;
  const float* w2r = W2 + (size_t)c * 512;
  for (int k = 0; k < 512; ++k) {
    float rk = rr[k], ak = ar[k];
    s  += wsr[k] * rk + wsr[512 + k] * ak;
    t1 += w1r[k] * rk;
    t2 += w2r[k] * rk;
  }
  orow[1026 + c] = fmaxf(s, 0.f);
  tmp1[i] = t1;
  tmp2[i] = t2;
}

// Head part 2: cos, euc, ntn.  grid = 32 (per row), block = 64.
__global__ void __launch_bounds__(64) head2_kernel(
    const float* __restrict__ a, const float* __restrict__ r,
    const float* __restrict__ tmp1, const float* __restrict__ tmp2,
    const float* __restrict__ Wv, const float* __restrict__ bv,
    float* __restrict__ out) {
  __shared__ float red[8][64];
  const int n = blockIdx.x, tid = threadIdx.x;
  const float* ar = a + n * 512;
  const float* rr = r + n * 512;
  float s_ar = 0, s_aa = 0, s_rr = 0, s_ee = 0, s_o1 = 0, s_o2 = 0, wv0 = 0, wv1 = 0;
  for (int k = tid; k < 512; k += 64) {
    float av = ar[k], rv = rr[k];
    s_ar += av * rv; s_aa += av * av; s_rr += rv * rv;
    float d = av - rv + 1e-6f;
    s_ee += d * d;
    s_o1 += tmp1[n * 512 + k] * av;
    s_o2 += tmp2[n * 512 + k] * av;
  }
  for (int k = tid; k < 1024; k += 64) {      // cat = [r, a]
    float cv = (k < 512) ? rr[k] : ar[k - 512];
    wv0 += Wv[k] * cv;
    wv1 += Wv[1024 + k] * cv;
  }
  red[0][tid] = s_ar; red[1][tid] = s_aa; red[2][tid] = s_rr; red[3][tid] = s_ee;
  red[4][tid] = s_o1; red[5][tid] = s_o2; red[6][tid] = wv0; red[7][tid] = wv1;
  __syncthreads();
  if (tid == 0) {
    float acc[8];
    for (int j = 0; j < 8; ++j) {
      float s = 0.f;
      for (int x = 0; x < 64; ++x) s += red[j][x];
      acc[j] = s;
    }
    float na = fmaxf(sqrtf(acc[1]), 1e-6f);
    float nr = fmaxf(sqrtf(acc[2]), 1e-6f);
    float* orow = out + (size_t)n * 1540;
    orow[1024] = acc[0] / (na * nr);
    orow[1025] = sqrtf(acc[3]);
    orow[1538] = fmaxf(acc[4] + acc[6] + bv[0], 0.f);
    orow[1539] = fmaxf(acc[5] + acc[7] + bv[1], 0.f);
  }
}

// ---------------------------------------------------------------------------
extern "C" void kernel_launch(void* const* d_in, const int* in_sizes, int n_in,
                              void* d_out, int out_size, void* d_ws, size_t ws_size,
                              hipStream_t stream) {
  (void)in_sizes; (void)n_in; (void)out_size; (void)ws_size;

  const int*   added   = (const int*)d_in[0];
  const int*   removed = (const int*)d_in[1];
  const float* h0_sent = (const float*)d_in[3];
  const float* h0_word = (const float*)d_in[4];
  const float* embed   = (const float*)d_in[5];
  const float* wW[11]; for (int i = 0; i < 11; ++i) wW[i] = (const float*)d_in[6 + i];
  const float* sW[11]; for (int i = 0; i < 11; ++i) sW[i] = (const float*)d_in[17 + i];
  const float* Ws  = (const float*)d_in[28]; const float* bs_ = (const float*)d_in[29];
  const float* W1  = (const float*)d_in[30]; const float* b1  = (const float*)d_in[31];
  const float* W2  = (const float*)d_in[32]; const float* b2  = (const float*)d_in[33];
  const float* Wv  = (const float*)d_in[34]; const float* bv  = (const float*)d_in[35];

  char*  wsb = (char*)d_ws;
  size_t off = 0;
  auto alloc = [&](size_t bytes) -> void* {
    off = (off + 255) & ~(size_t)255;
    void* p = wsb + off;
    off += bytes;
    return p;
  };

  const size_t SZ_WIH = (size_t)512 * 768 * sizeof(bf16);
  const size_t SZ_WHH = (size_t)256 * 768 * sizeof(bf16);
  const size_t SZ_WAC = (size_t)512 * 512 * sizeof(bf16);
  const size_t ROWS_W = (size_t)64 * 32 * 32;   // 65536 word rows
  const size_t ROWS_S = (size_t)8 * 8 * 32;     // 2048 sentence rows

  bf16* pw_wih_f = (bf16*)alloc(SZ_WIH);
  bf16* pw_whh_f = (bf16*)alloc(SZ_WHH);
  bf16* pw_wih_b = (bf16*)alloc(SZ_WIH);
  bf16* pw_whh_b = (bf16*)alloc(SZ_WHH);
  bf16* pw_wa    = (bf16*)alloc(SZ_WAC);
  bf16* pw_wc    = (bf16*)alloc(SZ_WAC);
  bf16* ps_wih_f = (bf16*)alloc(SZ_WIH);
  bf16* ps_whh_f = (bf16*)alloc(SZ_WHH);
  bf16* ps_wih_b = (bf16*)alloc(SZ_WIH);
  bf16* ps_whh_b = (bf16*)alloc(SZ_WHH);
  bf16* ps_wa    = (bf16*)alloc(SZ_WAC);
  bf16* ps_wc    = (bf16*)alloc(SZ_WAC);

  bf16*  xg       = (bf16*)alloc(ROWS_W * 512 * sizeof(bf16));  // 64 MB
  bf16*  wout     = (bf16*)alloc(ROWS_W * 512 * sizeof(bf16));  // 64 MB
  bf16*  scores_w = xg;  // word input fully consumed before scoring
  float* sents_f  = (float*)alloc((size_t)64 * 32 * 512 * sizeof(float));
  bf16*  sents_bf = (bf16*)alloc((size_t)64 * 32 * 512 * sizeof(bf16));
  bf16*  sout     = (bf16*)alloc(ROWS_S * 512 * sizeof(bf16));
  bf16*  scores_s = (bf16*)alloc(ROWS_S * 512 * sizeof(bf16));
  float* hunks    = (float*)alloc((size_t)8 * 32 * 512 * sizeof(float));
  float* avec     = (float*)alloc((size_t)32 * 512 * sizeof(float));
  float* rvec     = (float*)alloc((size_t)32 * 512 * sizeof(float));
  float* tmp1     = (float*)alloc((size_t)32 * 512 * sizeof(float));
  float* tmp2     = (float*)alloc((size_t)32 * 512 * sizeof(float));

  auto pack = [&](const float* W, bf16* dst, int K, int N) {
    pack_b_kernel<<<dim3((K / 32) * (N / 16)), dim3(32), 0, stream>>>(W, dst, K, N);
  };
  pack(wW[0], pw_wih_f, 512, 768);  pack(wW[1], pw_whh_f, 256, 768);
  pack(wW[4], pw_wih_b, 512, 768);  pack(wW[5], pw_whh_b, 256, 768);
  pack(wW[8], pw_wa, 512, 512);     pack(wW[10], pw_wc, 512, 512);
  pack(sW[0], ps_wih_f, 512, 768);  pack(sW[1], ps_whh_f, 256, 768);
  pack(sW[4], ps_wih_b, 512, 768);  pack(sW[5], ps_whh_b, 256, 768);
  pack(sW[8], ps_wa, 512, 512);     pack(sW[10], ps_wc, 512, 512);

  auto run_side = [&](const int* tok, float* vec) {
    gather_embed_kernel<<<65536, 128, 0, stream>>>(tok, embed, xg);
    bigru_scan_kernel<<<128, 256, 0, stream>>>(
        xg, h0_word, pw_wih_f, pw_whh_f, wW[2], wW[3],
        pw_wih_b, pw_whh_b, wW[6], wW[7], wout, 32, 512);
    attn_scores_kernel<<<2048, 256, 0, stream>>>(wout, pw_wa, wW[9], pw_wc, scores_w);
    softmax_pool_kernel<<<64, 256, 0, stream>>>(scores_w, wout, sents_f, sents_bf, 32);
    bigru_scan_kernel<<<16, 256, 0, stream>>>(
        sents_bf, h0_sent, ps_wih_f, ps_whh_f, sW[2], sW[3],
        ps_wih_b, ps_whh_b, sW[6], sW[7], sout, 8, 512);
    attn_scores_kernel<<<64, 256, 0, stream>>>(sout, ps_wa, sW[9], ps_wc, scores_s);
    softmax_pool_kernel<<<8, 256, 0, stream>>>(scores_s, sout, hunks, nullptr, 8);
    mean8_kernel<<<64, 256, 0, stream>>>(hunks, vec);
  };

  run_side(added, avec);
  run_side(removed, rvec);

  head1_kernel<<<64, 256, 0, stream>>>(avec, rvec, Ws, bs_, W1, b1, W2, b2,
                                       tmp1, tmp2, (float*)d_out);
  head2_kernel<<<32, 64, 0, stream>>>(avec, rvec, tmp1, tmp2, Wv, bv,
                                      (float*)d_out);
}
